// MultiHeadAttention_61349312856437
// MI455X (gfx1250) — matrix operand order repaired and checked by reading, not compiled, not run on hardware
//
#include <hip/hip_runtime.h>
#include <hip/hip_bf16.h>

#define S_LEN  2048
#define BATCH  2
#define DMODEL 1024
#define NHEAD  16
#define DHEAD  64
#define NROW   (S_LEN * BATCH)   // 4096 rows for the projection GEMMs

typedef __attribute__((ext_vector_type(16))) __bf16 v16bf;
typedef __attribute__((ext_vector_type(8)))  float  v8f;

// 16 bf16 = 32 bytes = one WMMA A/B operand (8 VGPRs)
union Frag16 {
    v16bf          v;
    unsigned short us[16];
    uint4          q[2];
};

__device__ __forceinline__ unsigned short f2bf(float f) {
    unsigned int u = __builtin_bit_cast(unsigned int, f);
    u += 0x7FFFu + ((u >> 16) & 1u);          // round-to-nearest-even
    return (unsigned short)(u >> 16);
}

__device__ __forceinline__ v8f zero8() {
    v8f z;
#pragma unroll
    for (int i = 0; i < 8; ++i) z[i] = 0.0f;
    return z;
}

__device__ __forceinline__ v8f wmma_bf16(const Frag16& a, const Frag16& b, v8f c) {
    return __builtin_amdgcn_wmma_f32_16x16x32_bf16(
        /*neg_a=*/false, a.v, /*neg_b=*/false, b.v,
        /*c_mod=*/(short)0, c, /*reuse_a=*/false, /*reuse_b=*/false);
}

// CDNA5 async memory->LDS copy, 16B per lane, tracked by ASYNCcnt.
__device__ __forceinline__ void async_load_b128(unsigned lds_off, const void* gptr) {
    asm volatile("global_load_async_to_lds_b128 %0, %1, off"
                 :: "v"(lds_off), "v"((unsigned long long)(size_t)gptr)
                 : "memory");
}
__device__ __forceinline__ void wait_async0() {
    asm volatile("s_wait_asynccnt 0" ::: "memory");
}

// ---------------------------------------------------------------------------
// Kernel 0: one-pass f32 -> bf16 conversion (activations and weights), so the
// GEMM inner loops are pure global_load_b128 + v_wmma with no VALU convert.
// ---------------------------------------------------------------------------
__global__ __launch_bounds__(256) void cvt_kernel(
    const float* __restrict__ src, unsigned short* __restrict__ dst, int n8)
{
    const int i = blockIdx.x * 256 + threadIdx.x;
    if (i >= n8) return;
    const float* s = src + (size_t)i * 8;
    float4 f0 = *(const float4*)(s);
    float4 f1 = *(const float4*)(s + 4);
    union { unsigned short us[8]; uint4 q; } o;
    o.us[0] = f2bf(f0.x); o.us[1] = f2bf(f0.y); o.us[2] = f2bf(f0.z); o.us[3] = f2bf(f0.w);
    o.us[4] = f2bf(f1.x); o.us[5] = f2bf(f1.y); o.us[6] = f2bf(f1.z); o.us[7] = f2bf(f1.w);
    *(uint4*)(dst + (size_t)i * 8) = o.q;
}

// ---------------------------------------------------------------------------
// Kernel 1: Q/K/V projections.  Y = X @ W^T + b, X:[4096,1024] bf16 row-major.
// blockIdx.z selects the projection.  Each wave computes a 32(M) x 64(N) tile.
// Outputs (bf16):
//   p=0: qh[(b*H+h)*S + s][dk]      p=1: kh same layout
//   p=2: vht[(b*H+h)*64 + dk][s]   (transposed: B-operand for P@V is contiguous)
// ---------------------------------------------------------------------------
__global__ __launch_bounds__(128) void proj_kernel(
    const unsigned short* __restrict__ xq, const unsigned short* __restrict__ xk,
    const unsigned short* __restrict__ xv,
    const unsigned short* __restrict__ wqb, const unsigned short* __restrict__ wkb,
    const unsigned short* __restrict__ wvb,
    const float* __restrict__ bqp, const float* __restrict__ bkp,
    const float* __restrict__ bvp,
    unsigned short* __restrict__ qh, unsigned short* __restrict__ kh,
    unsigned short* __restrict__ vht)
{
    const int p = blockIdx.z;
    const unsigned short* x  = (p == 0) ? xq  : (p == 1) ? xk  : xv;
    const unsigned short* w  = (p == 0) ? wqb : (p == 1) ? wkb : wvb;
    const float*          bs = (p == 0) ? bqp : (p == 1) ? bkp : bvp;

    const int wave = threadIdx.x >> 5;
    const int lane = threadIdx.x & 31;
    const int nn   = lane & 15;      // N column within 16x16 tile / A row
    const int hf   = lane >> 4;      // lane half (K-half selector)

    const int m0 = blockIdx.x * 32;                 // 128 M-tiles of 32 rows
    const int n0 = (blockIdx.y * 4 + wave) * 64;    // 16 N-groups of 64

    v8f acc[2][4];
#pragma unroll
    for (int mt = 0; mt < 2; ++mt)
#pragma unroll
        for (int nt = 0; nt < 4; ++nt) acc[mt][nt] = zero8();

    const unsigned short* arow0 = x + (size_t)(m0 + nn) * DMODEL;
    const unsigned short* arow1 = x + (size_t)(m0 + 16 + nn) * DMODEL;

    for (int k0 = 0; k0 < DMODEL; k0 += 32) {
        Frag16 a0, a1;
        // A 16-bit 16x32 layout: half 0 holds K {0..7,16..23}, half 1 {8..15,24..31}
        a0.q[0] = *(const uint4*)(arow0 + k0 + hf * 8);
        a0.q[1] = *(const uint4*)(arow0 + k0 + 16 + hf * 8);
        a1.q[0] = *(const uint4*)(arow1 + k0 + hf * 8);
        a1.q[1] = *(const uint4*)(arow1 + k0 + 16 + hf * 8);
        __builtin_prefetch(arow0 + k0 + 64, 0, 1);
        __builtin_prefetch(arow1 + k0 + 64, 0, 1);
#pragma unroll
        for (int nt = 0; nt < 4; ++nt) {
            const int ncol = n0 + nt * 16 + nn;
            Frag16 b;   // B[k][n] = W[n][k]: half h holds K {8h..8h+7, 16+8h..16+8h+7}
            const unsigned short* wrow = w + (size_t)ncol * DMODEL + k0 + hf * 8;
            b.q[0] = *(const uint4*)(wrow);
            b.q[1] = *(const uint4*)(wrow + 16);
            acc[0][nt] = wmma_bf16(a0, b, acc[0][nt]);
            acc[1][nt] = wmma_bf16(a1, b, acc[1][nt]);
        }
    }

#pragma unroll
    for (int nt = 0; nt < 4; ++nt) {
        const int   ncol = n0 + nt * 16 + nn;
        const float bias = bs[ncol];
        const int   h    = ncol >> 6;
        const int   dk   = ncol & 63;
#pragma unroll
        for (int mt = 0; mt < 2; ++mt) {
#pragma unroll
            for (int r = 0; r < 8; ++r) {
                const int row = m0 + mt * 16 + r + 8 * hf; // C-layout: M = vgpr + 8*half
                const int s   = row >> 1;                  // row = s*B + b, B==2
                const int bi  = row & 1;
                const int bh  = bi * NHEAD + h;
                const unsigned short val = f2bf(acc[mt][nt][r] + bias);
                if (p == 0)      qh [((size_t)bh * S_LEN + s) * DHEAD + dk] = val;
                else if (p == 1) kh [((size_t)bh * S_LEN + s) * DHEAD + dk] = val;
                else             vht[((size_t)bh * DHEAD + dk) * S_LEN + s] = val;
            }
        }
    }
}

// ---------------------------------------------------------------------------
// Kernel 2: flash-attention.  Block = 4 waves sharing one (b,h); each wave
// owns a 16-row query tile.  K/V tiles for each 32-key block are staged into
// LDS ONCE per block with CDNA5 async global->LDS copies (ASYNCcnt), then all
// 4 waves read WMMA operands from LDS.  Online softmax in base-2 domain; P is
// bounced through LDS (C-layout -> A-layout) for the PV WMMAs.
// ---------------------------------------------------------------------------
__global__ __launch_bounds__(128) void attn_kernel(
    const unsigned short* __restrict__ qh, const unsigned short* __restrict__ kh,
    const unsigned short* __restrict__ vht, const float* __restrict__ mask,
    unsigned short* __restrict__ ctx)
{
    // K tile: 32 rows(t) x 64(dk) bf16, row stride 72 elem (144B) -> low-conflict b128 reads
    __shared__ __align__(16) unsigned short kbuf[32 * 72];
    // V tile: 64 rows(dk) x 32(t) bf16, row stride 40 elem (80B)
    __shared__ __align__(16) unsigned short vbuf[64 * 40];
    // per-wave P staging: 16 x 32 bf16, row stride 40
    __shared__ __align__(16) unsigned short plds[4][16 * 40];

    const int tid  = threadIdx.x;
    const int wave = tid >> 5;
    const int lane = tid & 31;
    const int nn   = lane & 15;
    const int hf   = lane >> 4;

    const int bh = blockIdx.x >> 5;                    // (b*H + h); 32 blocks each
    const int q0 = (blockIdx.x & 31) * 64 + wave * 16; // wave-private query tile

    const unsigned kls = (unsigned)(size_t)(const void*)kbuf;
    const unsigned vls = (unsigned)(size_t)(const void*)vbuf;

    // Q tile as two A operands (dk 0..31 and 32..63)
    Frag16 aq0, aq1;
    {
        const unsigned short* qrow = qh + ((size_t)bh * S_LEN + q0 + nn) * DHEAD;
        aq0.q[0] = *(const uint4*)(qrow + hf * 8);
        aq0.q[1] = *(const uint4*)(qrow + 16 + hf * 8);
        aq1.q[0] = *(const uint4*)(qrow + 32 + hf * 8);
        aq1.q[1] = *(const uint4*)(qrow + 48 + hf * 8);
    }

    float mrow[8], lrow[8];
#pragma unroll
    for (int r = 0; r < 8; ++r) { mrow[r] = -1e30f; lrow[r] = 0.0f; }
    v8f o[4];
#pragma unroll
    for (int d = 0; d < 4; ++d) o[d] = zero8();

    const float L2E = 1.4426950408889634f;
    const float SC  = 0.125f * L2E;          // 1/sqrt(64) folded with log2(e)

    unsigned short* pl = plds[wave];
    const unsigned short* ksrc0 = kh  + (size_t)bh * S_LEN * DHEAD;
    const unsigned short* vsrc0 = vht + (size_t)bh * DHEAD * S_LEN;

    for (int t0 = 0; t0 < S_LEN; t0 += 32) {
        // make sure every wave finished reading the previous K/V tiles
        __syncthreads();

        // ---- async-stage K tile (contiguous 4KB) and V tile into LDS
        {
            const char* ks = (const char*)(ksrc0 + (size_t)t0 * DHEAD);
#pragma unroll
            for (int j = 0; j < 2; ++j) {
                const int idx = tid + j * 128;            // 16B chunk id, 0..255
                const int row = idx >> 3;                 // t row (128B each)
                const int cb  = (idx & 7) * 16;           // byte col
                async_load_b128(kls + (unsigned)(row * 144 + cb), ks + idx * 16);
            }
            const char* vs = (const char*)(vsrc0 + t0);
#pragma unroll
            for (int j = 0; j < 2; ++j) {
                const int idx = tid + j * 128;
                const int row = idx >> 2;                 // dk row (64B slice each)
                const int cb  = (idx & 3) * 16;
                async_load_b128(vls + (unsigned)(row * 80 + cb),
                                vs + (size_t)row * (S_LEN * 2) + cb);
            }
        }
        wait_async0();
        __syncthreads();

        // ---- scores: two 16x16 tiles (key cols t0..+15, t0+16..+31) from LDS
        v8f sc0 = zero8(), sc1 = zero8();
        {
            const unsigned short* kr = kbuf + (size_t)nn * 72 + hf * 8;
            Frag16 b0, b1;
            b0.q[0] = *(const uint4*)(kr);      b0.q[1] = *(const uint4*)(kr + 16);
            b1.q[0] = *(const uint4*)(kr + 32); b1.q[1] = *(const uint4*)(kr + 48);
            sc0 = wmma_bf16(aq0, b0, sc0);
            sc0 = wmma_bf16(aq1, b1, sc0);
        }
        {
            const unsigned short* kr = kbuf + (size_t)(16 + nn) * 72 + hf * 8;
            Frag16 b0, b1;
            b0.q[0] = *(const uint4*)(kr);      b0.q[1] = *(const uint4*)(kr + 16);
            b1.q[0] = *(const uint4*)(kr + 32); b1.q[1] = *(const uint4*)(kr + 48);
            sc1 = wmma_bf16(aq0, b0, sc1);
            sc1 = wmma_bf16(aq1, b1, sc1);
        }

        // ---- scale + mask (base-2 domain)
#pragma unroll
        for (int r = 0; r < 8; ++r) {
            const float* mr = mask + (size_t)(q0 + r + 8 * hf) * S_LEN + t0;
            sc0[r] = sc0[r] * SC + mr[nn]      * L2E;
            sc1[r] = sc1[r] * SC + mr[16 + nn] * L2E;
        }

        // ---- row max across the 16 lanes holding this row
        float tmax[8];
#pragma unroll
        for (int r = 0; r < 8; ++r) tmax[r] = fmaxf(sc0[r], sc1[r]);
#pragma unroll
        for (int off = 1; off < 16; off <<= 1)
#pragma unroll
            for (int r = 0; r < 8; ++r)
                tmax[r] = fmaxf(tmax[r], __shfl_xor(tmax[r], off, 32));

        float alpha[8];
#pragma unroll
        for (int r = 0; r < 8; ++r) {
            const float mnew = fmaxf(mrow[r], tmax[r]);
            alpha[r] = exp2f(mrow[r] - mnew);
            mrow[r]  = mnew;
        }

        // ---- exponentials + row sums
        float rsum[8];
#pragma unroll
        for (int r = 0; r < 8; ++r) {
            const float p0 = exp2f(sc0[r] - mrow[r]);
            const float p1 = exp2f(sc1[r] - mrow[r]);
            sc0[r] = p0; sc1[r] = p1;
            rsum[r] = p0 + p1;
        }
#pragma unroll
        for (int off = 1; off < 16; off <<= 1)
#pragma unroll
            for (int r = 0; r < 8; ++r)
                rsum[r] += __shfl_xor(rsum[r], off, 32);
#pragma unroll
        for (int r = 0; r < 8; ++r) lrow[r] = lrow[r] * alpha[r] + rsum[r];

        // ---- rescale running context
#pragma unroll
        for (int d = 0; d < 4; ++d)
#pragma unroll
            for (int r = 0; r < 8; ++r) o[d][r] *= alpha[r];

        // ---- P: C-layout regs -> LDS -> A-layout regs (bf16), per-wave
#pragma unroll
        for (int r = 0; r < 8; ++r) {
            pl[(r + 8 * hf) * 40 + nn]      = f2bf(sc0[r]);
            pl[(r + 8 * hf) * 40 + 16 + nn] = f2bf(sc1[r]);
        }
        __builtin_amdgcn_wave_barrier();
        asm volatile("s_wait_dscnt 0" ::: "memory");   // DS ops in-order per wave
        Frag16 ap;
        {
            const unsigned short* prow = pl + nn * 40;
            ap.q[0] = *(const uint4*)(prow + hf * 8);
            ap.q[1] = *(const uint4*)(prow + 16 + hf * 8);
        }
        __builtin_amdgcn_wave_barrier();
        asm volatile("s_wait_dscnt 0" ::: "memory");

        // ---- ctx += P @ V  (Vt tile in LDS: K = key index is contiguous)
#pragma unroll
        for (int d = 0; d < 4; ++d) {
            const unsigned short* vr = vbuf + (size_t)(d * 16 + nn) * 40 + hf * 8;
            Frag16 bv;
            bv.q[0] = *(const uint4*)(vr);
            bv.q[1] = *(const uint4*)(vr + 16);
            o[d] = wmma_bf16(ap, bv, o[d]);
        }
    }

    // ---- normalize + write combined-head ctx (bf16) as [S*B, D] rows
    const int b = bh >> 4;
    const int h = bh & 15;
#pragma unroll
    for (int r = 0; r < 8; ++r) {
        const float inv = 1.0f / lrow[r];
        const int   s   = q0 + r + 8 * hf;
#pragma unroll
        for (int d = 0; d < 4; ++d) {
            ctx[((size_t)(s * BATCH + b)) * DMODEL + h * DHEAD + d * 16 + nn] =
                f2bf(o[d][r] * inv);
        }
    }
}

// ---------------------------------------------------------------------------
// Kernel 3: output projection.  out = ctx(bf16) @ w_o^T + b_o, f32 out.
// Each wave computes a 32(M) x 64(N) tile.
// ---------------------------------------------------------------------------
__global__ __launch_bounds__(128) void out_kernel(
    const unsigned short* __restrict__ ctx, const unsigned short* __restrict__ wob,
    const float* __restrict__ bo, float* __restrict__ out)
{
    const int wave = threadIdx.x >> 5;
    const int lane = threadIdx.x & 31;
    const int nn   = lane & 15;
    const int hf   = lane >> 4;

    const int m0 = blockIdx.x * 32;
    const int n0 = (blockIdx.y * 4 + wave) * 64;

    v8f acc[2][4];
#pragma unroll
    for (int mt = 0; mt < 2; ++mt)
#pragma unroll
        for (int nt = 0; nt < 4; ++nt) acc[mt][nt] = zero8();

    const unsigned short* arow0 = ctx + (size_t)(m0 + nn) * DMODEL;
    const unsigned short* arow1 = ctx + (size_t)(m0 + 16 + nn) * DMODEL;

    for (int k0 = 0; k0 < DMODEL; k0 += 32) {
        Frag16 a0, a1;
        a0.q[0] = *(const uint4*)(arow0 + k0 + hf * 8);
        a0.q[1] = *(const uint4*)(arow0 + k0 + 16 + hf * 8);
        a1.q[0] = *(const uint4*)(arow1 + k0 + hf * 8);
        a1.q[1] = *(const uint4*)(arow1 + k0 + 16 + hf * 8);
        __builtin_prefetch(arow0 + k0 + 64, 0, 1);
        __builtin_prefetch(arow1 + k0 + 64, 0, 1);
#pragma unroll
        for (int nt = 0; nt < 4; ++nt) {
            const int ncol = n0 + nt * 16 + nn;
            Frag16 b;
            const unsigned short* wrow = wob + (size_t)ncol * DMODEL + k0 + hf * 8;
            b.q[0] = *(const uint4*)(wrow);
            b.q[1] = *(const uint4*)(wrow + 16);
            acc[0][nt] = wmma_bf16(a0, b, acc[0][nt]);
            acc[1][nt] = wmma_bf16(a1, b, acc[1][nt]);
        }
    }

#pragma unroll
    for (int nt = 0; nt < 4; ++nt) {
        const int   ncol = n0 + nt * 16 + nn;
        const float bias = bo[ncol];
#pragma unroll
        for (int mt = 0; mt < 2; ++mt)
#pragma unroll
            for (int r = 0; r < 8; ++r)
                out[(size_t)(m0 + mt * 16 + r + 8 * hf) * DMODEL + ncol] =
                    acc[mt][nt][r] + bias;
    }
}

// ---------------------------------------------------------------------------
extern "C" void kernel_launch(void* const* d_in, const int* in_sizes, int n_in,
                              void* d_out, int out_size, void* d_ws, size_t ws_size,
                              hipStream_t stream) {
    const float* q    = (const float*)d_in[0];
    const float* k    = (const float*)d_in[1];
    const float* v    = (const float*)d_in[2];
    const float* mask = (const float*)d_in[3];
    const float* wq   = (const float*)d_in[4];
    const float* bq   = (const float*)d_in[5];
    const float* wk   = (const float*)d_in[6];
    const float* bk   = (const float*)d_in[7];
    const float* wv   = (const float*)d_in[8];
    const float* bv   = (const float*)d_in[9];
    const float* wo   = (const float*)d_in[10];
    const float* bo   = (const float*)d_in[11];
    float* out = (float*)d_out;

    const size_t xel = (size_t)NROW * DMODEL;                 // 4M elements
    const size_t wel = (size_t)DMODEL * DMODEL;               // 1M elements
    const size_t hel = (size_t)BATCH * NHEAD * S_LEN * DHEAD; // 4M elements

    unsigned short* p   = (unsigned short*)d_ws;
    unsigned short* xqb = p;  p += xel;
    unsigned short* xkb = p;  p += xel;
    unsigned short* xvb = p;  p += xel;
    unsigned short* wqb = p;  p += wel;
    unsigned short* wkb = p;  p += wel;
    unsigned short* wvb = p;  p += wel;
    unsigned short* wob = p;  p += wel;
    unsigned short* qh  = p;  p += hel;
    unsigned short* kh  = p;  p += hel;
    unsigned short* vht = p;  p += hel;
    unsigned short* ctx = p;  p += hel;   // total = 32M ushorts = 64 MB

    dim3 blk(128);   // 4 wave32 waves per block

    // one-pass bf16 conversion of activations and weights
    const int xn8 = (int)(xel / 8), wn8 = (int)(wel / 8);
    cvt_kernel<<<dim3((xn8 + 255) / 256), dim3(256), 0, stream>>>(q,  xqb, xn8);
    cvt_kernel<<<dim3((xn8 + 255) / 256), dim3(256), 0, stream>>>(k,  xkb, xn8);
    cvt_kernel<<<dim3((xn8 + 255) / 256), dim3(256), 0, stream>>>(v,  xvb, xn8);
    cvt_kernel<<<dim3((wn8 + 255) / 256), dim3(256), 0, stream>>>(wq, wqb, wn8);
    cvt_kernel<<<dim3((wn8 + 255) / 256), dim3(256), 0, stream>>>(wk, wkb, wn8);
    cvt_kernel<<<dim3((wn8 + 255) / 256), dim3(256), 0, stream>>>(wv, wvb, wn8);
    cvt_kernel<<<dim3((wn8 + 255) / 256), dim3(256), 0, stream>>>(wo, wob, wn8);

    proj_kernel<<<dim3(NROW / 32, 4, 3), blk, 0, stream>>>(
        xqb, xkb, xvb, wqb, wkb, wvb, bq, bk, bv, qh, kh, vht);
    attn_kernel<<<dim3((BATCH * NHEAD) * (S_LEN / 64)), blk, 0, stream>>>(
        qh, kh, vht, mask, ctx);
    out_kernel<<<dim3(NROW / 32, 4), blk, 0, stream>>>(ctx, wob, bo, out);

    (void)in_sizes; (void)n_in; (void)out_size; (void)ws_size;
}
